// RetinaFaceDetector_19911468384920
// MI455X (gfx1250) — compile-verified, matrix operand order checked by reading, not run on hardware
//
#include <hip/hip_runtime.h>
#include <hip/hip_bf16.h>
#include <stdint.h>

// ---------------------------------------------------------------------------
// RetinaFace post-processing for MI455X (gfx1250, wave32, WMMA).
// decode -> keyed bitonic sort -> gather top-5000 -> tiled greedy NMS with
// WMMA-based suppression reduction (boxes staged to LDS with
// global_load_async_to_lds) -> stable compaction to 750 rows.
// ---------------------------------------------------------------------------

typedef __attribute__((ext_vector_type(16))) _Float16 v16h;
typedef __attribute__((ext_vector_type(8)))  _Float16 v8h;
typedef __attribute__((ext_vector_type(8)))  float    v8f;
typedef __attribute__((ext_vector_type(4)))  float    v4f;

namespace cfg {
constexpr int B     = 4;
constexpr int N     = 16800;
constexpr int M     = 32768;   // sort size (pow2 >= N)
constexpr int TOPK  = 5000;
constexpr int K     = 5008;    // TOPK padded to multiple of 16
constexpr int KEEP  = 750;
constexpr float VAR0 = 0.1f;
constexpr float VAR1 = 0.2f;
constexpr float CONF = 0.5f;
constexpr float NMS  = 0.4f;
constexpr float IMG  = 640.0f;

// LDS layout for the NMS kernel (dynamic shared segment, base offset 0)
constexpr unsigned OFF_BOX   = 0;                        // K * float4
constexpr unsigned OFF_AR    = OFF_BOX   + (unsigned)K * 16u;  // K * float
constexpr unsigned OFF_PSUM  = OFF_AR    + (unsigned)K * 4u;   // 16 * float
constexpr unsigned OFF_KEPTH = OFF_PSUM  + 64u;                // K * f16
constexpr unsigned OFF_KEPTB = OFF_KEPTH + (unsigned)K * 2u;   // K bytes
constexpr unsigned OFF_VALID = OFF_KEPTB + (unsigned)K;        // K bytes
constexpr unsigned OFF_SUPT  = OFF_VALID + (unsigned)K;        // 256 bytes
constexpr unsigned LDS_BYTES = OFF_SUPT + 256u;
}

// ---------------------------------------------------------------------------
// Kernel 1: decode boxes / landmarks / scores
// ---------------------------------------------------------------------------
__global__ void decode_kernel(const float* __restrict__ loc,
                              const float* __restrict__ conf,
                              const float* __restrict__ lmk,
                              const float* __restrict__ priors,
                              float* __restrict__ boxes,
                              float* __restrict__ lms,
                              float* __restrict__ scores) {
    const int i = blockIdx.x * blockDim.x + threadIdx.x;
    if (i >= cfg::B * cfg::N) return;
    const int n = i % cfg::N;

    // speculative prefetch ahead (emits global_prefetch_b8)
    __builtin_prefetch(&loc[(size_t)(i) * 4 + 4096], 0, 0);

    const float px = priors[n * 4 + 0], py = priors[n * 4 + 1];
    const float pw = priors[n * 4 + 2], ph = priors[n * 4 + 3];
    const float lx = loc[(size_t)i * 4 + 0], ly = loc[(size_t)i * 4 + 1];
    const float lw = loc[(size_t)i * 4 + 2], lh = loc[(size_t)i * 4 + 3];

    const float cx = px + lx * cfg::VAR0 * pw;
    const float cy = py + ly * cfg::VAR0 * ph;
    const float w  = pw * __expf(lw * cfg::VAR1);
    const float h  = ph * __expf(lh * cfg::VAR1);
    const float x1 = (cx - 0.5f * w) * cfg::IMG;
    const float y1 = (cy - 0.5f * h) * cfg::IMG;
    boxes[(size_t)i * 4 + 0] = x1;
    boxes[(size_t)i * 4 + 1] = y1;
    boxes[(size_t)i * 4 + 2] = x1 + w * cfg::IMG;
    boxes[(size_t)i * 4 + 3] = y1 + h * cfg::IMG;

#pragma unroll
    for (int p = 0; p < 5; ++p) {
        const float ax = lmk[(size_t)i * 10 + 2 * p + 0];
        const float ay = lmk[(size_t)i * 10 + 2 * p + 1];
        lms[(size_t)i * 10 + 2 * p + 0] = (px + ax * cfg::VAR0 * pw) * cfg::IMG;
        lms[(size_t)i * 10 + 2 * p + 1] = (py + ay * cfg::VAR0 * ph) * cfg::IMG;
    }
    scores[i] = conf[(size_t)i * 2 + 1];
}

// ---------------------------------------------------------------------------
// Kernel 2: build sortable keys. Ascending sort of key == stable descending
// argsort of masked score (invalid -> -inf, ties broken by ascending index).
// ---------------------------------------------------------------------------
__global__ void keyinit_kernel(const float* __restrict__ scores,
                               unsigned long long* __restrict__ keys) {
    const int t = blockIdx.x * blockDim.x + threadIdx.x;
    if (t >= cfg::B * cfg::M) return;
    const int b = t / cfg::M;
    const int n = t % cfg::M;
    unsigned long long key;
    if (n < cfg::N) {
        const float s = scores[b * cfg::N + n];
        const float m = (s > cfg::CONF) ? s : -__builtin_inff();
        unsigned u = __float_as_uint(m);
        u ^= (u >> 31) ? 0xFFFFFFFFu : 0x80000000u;  // totally-ordered ascending
        const unsigned d = ~u;                        // descending
        key = ((unsigned long long)d << 32) | (unsigned)n;
    } else {
        key = ~0ull;                                  // padding sorts last
    }
    keys[t] = key;
}

// ---------------------------------------------------------------------------
// Kernel 3: bitonic compare-exchange step (ascending), one step per launch.
// ---------------------------------------------------------------------------
__global__ void bitonic_kernel(unsigned long long* __restrict__ keys,
                               int k, int j) {
    const int t = blockIdx.x * blockDim.x + threadIdx.x;
    if (t >= cfg::B * cfg::M) return;
    const int b = t / cfg::M;
    const int i = t % cfg::M;
    const int ixj = i ^ j;
    if (ixj <= i) return;
    unsigned long long* kb = keys + (size_t)b * cfg::M;
    const unsigned long long a = kb[i];
    const unsigned long long c = kb[ixj];
    const bool up = ((i & k) == 0);
    if ((a > c) == up) { kb[i] = c; kb[ixj] = a; }
}

// ---------------------------------------------------------------------------
// Kernel 4: gather sorted top-K candidates (+ padding) into dense arrays.
// ---------------------------------------------------------------------------
__global__ void gather_kernel(const unsigned long long* __restrict__ keys,
                              const float* __restrict__ boxes,
                              const float* __restrict__ lms,
                              const float* __restrict__ scores,
                              float* __restrict__ sb,
                              float* __restrict__ sarea,
                              float* __restrict__ ss,
                              float* __restrict__ slm,
                              unsigned char* __restrict__ svalid) {
    const int t = blockIdx.x * blockDim.x + threadIdx.x;
    if (t >= cfg::B * cfg::K) return;
    const int b = t / cfg::K;
    const int i = t % cfg::K;
    if (i < cfg::TOPK) {
        const int idx = (int)(keys[(size_t)b * cfg::M + i] & 0xFFFFFFFFull);
        const float* bx = boxes + ((size_t)b * cfg::N + idx) * 4;
        const float x1 = bx[0], y1 = bx[1], x2 = bx[2], y2 = bx[3];
        sb[(size_t)t * 4 + 0] = x1;
        sb[(size_t)t * 4 + 1] = y1;
        sb[(size_t)t * 4 + 2] = x2;
        sb[(size_t)t * 4 + 3] = y2;
        sarea[t] = (x2 - x1 + 1.0f) * (y2 - y1 + 1.0f);
        const float s = scores[b * cfg::N + idx];
        ss[t] = s;
        svalid[t] = (s > cfg::CONF) ? 1 : 0;
#pragma unroll
        for (int c = 0; c < 10; ++c)
            slm[(size_t)t * 10 + c] = lms[((size_t)b * cfg::N + idx) * 10 + c];
    } else {
        sb[(size_t)t * 4 + 0] = 0.f; sb[(size_t)t * 4 + 1] = 0.f;
        sb[(size_t)t * 4 + 2] = 0.f; sb[(size_t)t * 4 + 3] = 0.f;
        sarea[t] = 1.0f;
        ss[t] = 0.f;
        svalid[t] = 0;
#pragma unroll
        for (int c = 0; c < 10; ++c) slm[(size_t)t * 10 + c] = 0.f;
    }
}

// ---------------------------------------------------------------------------
// Kernel 5: tiled greedy NMS. One wave32 per image. Candidate boxes + areas
// are staged into LDS with global_load_async_to_lds (ASYNCcnt), then the
// suppression-by-kept-prefix reduction runs on the WMMA pipe:
// A = sup bits (16 rows x 32 K, f16), B = kept flags replicated, acc f32.
// A-layout identity: element e<8 -> K = kbase0+e ; e>=8 -> K = 16+kbase0+e-8,
// so the B operand is two aligned 16-byte LDS vector loads of kept flags.
// Intra-tile (16-row) dependency resolved serially by lane 0.
// ---------------------------------------------------------------------------
__global__ __launch_bounds__(32)
void nms_kernel(const float* __restrict__ sb,
                const float* __restrict__ sarea,
                const unsigned char* __restrict__ svalid,
                unsigned char* __restrict__ kept_out) {
    extern __shared__ char smem[];
    const int b    = blockIdx.x;
    const int lane = threadIdx.x;

    v4f*   boxL  = (v4f*)(smem + cfg::OFF_BOX);
    float* ar    = (float*)(smem + cfg::OFF_AR);
    float* psum  = (float*)(smem + cfg::OFF_PSUM);
    _Float16* keptH = (_Float16*)(smem + cfg::OFF_KEPTH);
    unsigned char* keptB  = (unsigned char*)(smem + cfg::OFF_KEPTB);
    unsigned char* validB = (unsigned char*)(smem + cfg::OFF_VALID);
    unsigned char* supT   = (unsigned char*)(smem + cfg::OFF_SUPT);

    // ---- async stage: boxes (b128) and areas (b32) straight into LDS ------
    for (int i = lane; i < cfg::K; i += 32) {
        const unsigned ldsBox = cfg::OFF_BOX + (unsigned)i * 16u;
        const unsigned long long gBox =
            (unsigned long long)(uintptr_t)(sb + ((size_t)b * cfg::K + i) * 4);
        asm volatile("global_load_async_to_lds_b128 %0, %1, off"
                     :: "v"(ldsBox), "v"(gBox) : "memory");
        const unsigned ldsAr = cfg::OFF_AR + (unsigned)i * 4u;
        const unsigned long long gAr =
            (unsigned long long)(uintptr_t)(sarea + (size_t)b * cfg::K + i);
        asm volatile("global_load_async_to_lds_b32 %0, %1, off"
                     :: "v"(ldsAr), "v"(gAr) : "memory");
        keptH[i]  = (_Float16)0.0f;
        keptB[i]  = 0;
        validB[i] = svalid[b * cfg::K + i];
    }
    asm volatile("s_wait_asynccnt 0x0" ::: "memory");
    __syncthreads();

    const int kbase0 = (lane < 16) ? 0 : 8;   // K-half owned by this lane group
    const int row    = lane & 15;             // A-matrix row M for this lane

    for (int t = 0; t < cfg::K / 16; ++t) {
        const int r0 = t * 16;
        const int gi = r0 + row;
        const v4f   ib = boxL[gi];
        const float ia = ar[gi];

        // --- suppression count vs. all previously-decided candidates -------
        v8f acc = {};
#pragma unroll 2
        for (int jb = 0; jb < r0; jb += 32) {
            v16h a, bm;
#pragma unroll
            for (int e = 0; e < 8; ++e) {
                const int j = jb + kbase0 + e;
                const v4f bx = boxL[j];
                const float w = fmaxf(0.0f, fminf(ib[2], bx[2]) - fmaxf(ib[0], bx[0]) + 1.0f);
                const float h = fmaxf(0.0f, fminf(ib[3], bx[3]) - fmaxf(ib[1], bx[1]) + 1.0f);
                const float inter = w * h;
                const float iou = inter / (ia + ar[j] - inter);
                a[e] = (iou > cfg::NMS) ? (_Float16)1.0f : (_Float16)0.0f;
            }
#pragma unroll
            for (int e = 0; e < 8; ++e) {
                const int j = jb + 16 + kbase0 + e;
                const v4f bx = boxL[j];
                const float w = fmaxf(0.0f, fminf(ib[2], bx[2]) - fmaxf(ib[0], bx[0]) + 1.0f);
                const float h = fmaxf(0.0f, fminf(ib[3], bx[3]) - fmaxf(ib[1], bx[1]) + 1.0f);
                const float inter = w * h;
                const float iou = inter / (ia + ar[j] - inter);
                a[8 + e] = (iou > cfg::NMS) ? (_Float16)1.0f : (_Float16)0.0f;
            }
            // kept flags for the same K slots: two aligned 16B LDS loads
            const v8h lo = *(const v8h*)(smem + cfg::OFF_KEPTH + (unsigned)(jb + kbase0) * 2u);
            const v8h hi = *(const v8h*)(smem + cfg::OFF_KEPTH + (unsigned)(jb + 16 + kbase0) * 2u);
#pragma unroll
            for (int e = 0; e < 8; ++e) { bm[e] = lo[e]; bm[8 + e] = hi[e]; }

            acc = __builtin_amdgcn_wmma_f32_16x16x32_f16(
                false, a, false, bm, (short)0, acc, false, false);
        }

        // column 0 of D: lane 0 holds rows 0..7, lane 16 holds rows 8..15
        if (lane == 0) {
#pragma unroll
            for (int v = 0; v < 8; ++v) psum[v] = acc[v];
        }
        if (lane == 16) {
#pragma unroll
            for (int v = 0; v < 8; ++v) psum[8 + v] = acc[v];
        }

        // --- intra-tile 16x16 suppression bits ----------------------------
#pragma unroll
        for (int c0 = 0; c0 < 8; ++c0) {
            const int c = kbase0 + c0;
            const v4f bx = boxL[r0 + c];
            const float w = fmaxf(0.0f, fminf(ib[2], bx[2]) - fmaxf(ib[0], bx[0]) + 1.0f);
            const float h = fmaxf(0.0f, fminf(ib[3], bx[3]) - fmaxf(ib[1], bx[1]) + 1.0f);
            const float inter = w * h;
            const float iou = inter / (ia + ar[r0 + c] - inter);
            supT[row * 16 + c] = (iou > cfg::NMS) ? 1 : 0;
        }
        __syncthreads();

        // --- serial resolve of the 16-row dependency chain ----------------
        if (lane == 0) {
            unsigned keptMask = 0;
            for (int r = 0; r < 16; ++r) {
                bool keep = (validB[r0 + r] != 0) && (psum[r] < 0.5f);
                if (keep) {
                    for (int c = 0; c < r; ++c) {
                        if (((keptMask >> c) & 1u) && supT[r * 16 + c]) {
                            keep = false; break;
                        }
                    }
                }
                if (keep) keptMask |= (1u << r);
                keptH[r0 + r] = keep ? (_Float16)1.0f : (_Float16)0.0f;
                keptB[r0 + r] = keep ? 1 : 0;
            }
        }
        __syncthreads();
    }

    for (int i = lane; i < cfg::K; i += 32)
        kept_out[b * cfg::K + i] = keptB[i];
}

// ---------------------------------------------------------------------------
// Kernel 6: stable compaction of kept rows into 750x15 dets + mask.
// ---------------------------------------------------------------------------
__global__ __launch_bounds__(32)
void output_kernel(const float* __restrict__ sb,
                   const float* __restrict__ ss,
                   const float* __restrict__ slm,
                   const unsigned char* __restrict__ kept,
                   float* __restrict__ out) {
    const int b = blockIdx.x;
    const int lane = threadIdx.x;
    float* dets = out;                                   // [B][KEEP][15]
    float* mask = out + (size_t)cfg::B * cfg::KEEP * 15; // [B][KEEP]

    __shared__ int sel[cfg::KEEP];
    __shared__ int nk;

    for (int i = lane; i < cfg::KEEP * 15; i += 32)
        dets[(size_t)b * cfg::KEEP * 15 + i] = 0.0f;
    for (int i = lane; i < cfg::KEEP; i += 32)
        mask[b * cfg::KEEP + i] = 0.0f;

    if (lane == 0) {
        int rank = 0;
        for (int i = 0; i < cfg::TOPK; ++i) {
            if (kept[b * cfg::K + i]) {
                if (rank < cfg::KEEP) sel[rank] = i;
                ++rank;
            }
        }
        nk = (rank < cfg::KEEP) ? rank : cfg::KEEP;
    }
    __syncthreads();

    for (int r = lane; r < nk; r += 32) {
        const int i = sel[r];
        float* row = dets + ((size_t)b * cfg::KEEP + r) * 15;
        const float* bx = sb + ((size_t)b * cfg::K + i) * 4;
        row[0] = bx[0]; row[1] = bx[1]; row[2] = bx[2]; row[3] = bx[3];
        row[4] = ss[b * cfg::K + i];
#pragma unroll
        for (int c = 0; c < 10; ++c)
            row[5 + c] = slm[((size_t)b * cfg::K + i) * 10 + c];
        mask[b * cfg::KEEP + r] = 1.0f;
    }
}

// ---------------------------------------------------------------------------
// Host launcher
// ---------------------------------------------------------------------------
extern "C" void kernel_launch(void* const* d_in, const int* in_sizes, int n_in,
                              void* d_out, int out_size, void* d_ws, size_t ws_size,
                              hipStream_t stream) {
    (void)in_sizes; (void)n_in; (void)out_size; (void)ws_size;
    using namespace cfg;

    const float* loc    = (const float*)d_in[0];
    const float* conf   = (const float*)d_in[1];
    const float* landms = (const float*)d_in[2];
    const float* priors = (const float*)d_in[3];

    char* ws = (char*)d_ws;
    size_t off = 0;
    auto take = [&](size_t bytes) -> char* {
        off = (off + 255) & ~(size_t)255;
        char* p = ws + off;
        off += bytes;
        return p;
    };

    float* boxes  = (float*)take((size_t)B * N * 4 * sizeof(float));
    float* lms    = (float*)take((size_t)B * N * 10 * sizeof(float));
    float* scores = (float*)take((size_t)B * N * sizeof(float));
    unsigned long long* keys =
        (unsigned long long*)take((size_t)B * M * sizeof(unsigned long long));
    float* sbx    = (float*)take((size_t)B * K * 4 * sizeof(float));
    float* sarea  = (float*)take((size_t)B * K * sizeof(float));
    float* ssc    = (float*)take((size_t)B * K * sizeof(float));
    float* slm    = (float*)take((size_t)B * K * 10 * sizeof(float));
    unsigned char* svalid = (unsigned char*)take((size_t)B * K);
    unsigned char* kept   = (unsigned char*)take((size_t)B * K);

    // 1) decode
    decode_kernel<<<(B * N + 255) / 256, 256, 0, stream>>>(
        loc, conf, landms, priors, boxes, lms, scores);

    // 2) keys
    keyinit_kernel<<<(B * M + 255) / 256, 256, 0, stream>>>(scores, keys);

    // 3) bitonic sort (120 steps, keys stay hot in L2)
    for (int k = 2; k <= M; k <<= 1)
        for (int j = k >> 1; j >= 1; j >>= 1)
            bitonic_kernel<<<(B * M) / 256, 256, 0, stream>>>(keys, k, j);

    // 4) gather sorted top-K
    gather_kernel<<<(B * K + 255) / 256, 256, 0, stream>>>(
        keys, boxes, lms, scores, sbx, sarea, ssc, slm, svalid);

    // 5) WMMA-tiled greedy NMS, one wave per image; ~118 KB dynamic LDS of
    //    the 320 KB/WGP holds all candidate state.
    nms_kernel<<<B, 32, LDS_BYTES, stream>>>(sbx, sarea, svalid, kept);

    // 6) compact to output
    output_kernel<<<B, 32, 0, stream>>>(sbx, ssc, slm, kept, (float*)d_out);
}
